// MultiHeadAttention_3D_44100724195386
// MI455X (gfx1250) — matrix operand order repaired and checked by reading, not compiled, not run on hardware
//
#include <hip/hip_runtime.h>

typedef __attribute__((ext_vector_type(16))) _Float16 v16h;
typedef __attribute__((ext_vector_type(8)))  float    v8f;

union V16H {
    v16h h;
    uint4 u[2];
    _Float16 f[16];
};

#define TOK_TOTAL 65536   // B*H*W*T
#define DM 256            // model dim
#define TSEQ 64
#define NHEADS 8
#define DHEAD 32

// ---------------------------------------------------------------------------
// GEMM: C[M x 256] = X[M x 256] @ W[256 x 256] + bias
// Templated on input (f32/f16) and output (f16/f32) precision.
// Block: 256 threads (8 waves), 64 rows per block. Wave (mi,ng) computes
// rows [16*mi,+16) x cols [128*ng,+128) as 8 WMMA tiles. Double-buffered LDS:
// next chunk's global loads issue before the WMMAs, commit after.
// ---------------------------------------------------------------------------
template <bool IN_HALF, bool OUT_FLOAT>
__global__ __launch_bounds__(256) void gemm_d256(
    const void* __restrict__ Xv, const float* __restrict__ W,
    const float* __restrict__ bias, void* __restrict__ Cv)
{
    const float*    __restrict__ Xf = (const float*)Xv;
    const _Float16* __restrict__ Xh = (const _Float16*)Xv;
    float*    __restrict__ Cf = (float*)Cv;
    _Float16* __restrict__ Ch = (_Float16*)Cv;

    __shared__ __align__(16) _Float16 Xs[2][64 * 40];    // row stride 40 halves
    __shared__ __align__(16) _Float16 Ws[2][256 * 40];   // transposed: [n*40+k]

    const int tid  = threadIdx.x;
    const int lane = tid & 31;
    const int wave = tid >> 5;
    const int mi   = wave >> 1;      // 0..3  M sub-tile
    const int ng   = wave & 1;       // 0..1  N group of 128
    const int hf   = lane >> 4;
    const int l16  = lane & 15;
    const long rowBase = (long)blockIdx.x * 64;

    // staging assignments
    const int xr = (tid * 8) >> 5;          // 0..63
    const int xc = (tid * 8) & 31;          // 0,8,16,24
    const long xg = (rowBase + xr) * DM + xc;
    const int kp = tid >> 4;                // k pair = 2*kp (0..30)
    const int nb = (tid & 15) * 16;         // 0..240

    v8f acc[8];
    const v8f vzero = {0.f,0.f,0.f,0.f,0.f,0.f,0.f,0.f};
#pragma unroll
    for (int t = 0; t < 8; ++t) acc[t] = vzero;

    float xfl[8];
    uint4 xh4;
    float wst[2][16];

    // ---- prologue: stage chunk 0 into buffer 0 ----
    {
        if (IN_HALF) {
            xh4 = *(const uint4*)(Xh + xg);
        } else {
#pragma unroll
            for (int j = 0; j < 8; ++j) xfl[j] = Xf[xg + j];
        }
        const float* wp = W + (long)(2 * kp) * DM + nb;
#pragma unroll
        for (int j = 0; j < 16; ++j) { wst[0][j] = wp[j]; wst[1][j] = wp[DM + j]; }

        if (IN_HALF) {
            *(uint4*)&Xs[0][xr * 40 + xc] = xh4;
        } else {
            V16H tv;
#pragma unroll
            for (int j = 0; j < 8; ++j) tv.f[j] = (_Float16)xfl[j];
            *(uint4*)&Xs[0][xr * 40 + xc] = tv.u[0];
        }
#pragma unroll
        for (int j = 0; j < 16; ++j) {
            union { _Float16 h[2]; unsigned int u; } pk;
            pk.h[0] = (_Float16)wst[0][j];
            pk.h[1] = (_Float16)wst[1][j];
            *(unsigned int*)&Ws[0][(nb + j) * 40 + 2 * kp] = pk.u;
        }
    }
    __syncthreads();

    int buf = 0;
    for (int c = 0; c < 8; ++c) {
        // ---- issue next chunk's global loads (hidden behind WMMAs) ----
        if (c < 7) {
            const int k0 = (c + 1) * 32;
            if (IN_HALF) {
                xh4 = *(const uint4*)(Xh + xg + k0);
            } else {
#pragma unroll
                for (int j = 0; j < 8; ++j) xfl[j] = Xf[xg + k0 + j];
            }
            const float* wp = W + (long)(k0 + 2 * kp) * DM + nb;
#pragma unroll
            for (int j = 0; j < 16; ++j) { wst[0][j] = wp[j]; wst[1][j] = wp[DM + j]; }
        }

        // ---- compute from current buffer ----
        V16H a;   // lane row M = 16*mi+l16; K groups 8*hf+i, 16+8*hf+i
        {
            const _Float16* xrp = &Xs[buf][(16 * mi + l16) * 40];
            a.u[0] = *(const uint4*)(xrp + 8 * hf);
            a.u[1] = *(const uint4*)(xrp + 16 + 8 * hf);
        }
#pragma unroll
        for (int nt = 0; nt < 8; ++nt) {
            V16H bf;  // lane col N = 128*ng+16*nt+l16; K groups 8*hf+i, 16+8*hf+i
            const _Float16* wr = &Ws[buf][(128 * ng + 16 * nt + l16) * 40];
            bf.u[0] = *(const uint4*)(wr + 8 * hf);
            bf.u[1] = *(const uint4*)(wr + 16 + 8 * hf);
            acc[nt] = __builtin_amdgcn_wmma_f32_16x16x32_f16(
                false, a.h, false, bf.h, (short)0, acc[nt], false, false);
        }

        // ---- commit next chunk to the alternate buffer ----
        if (c < 7) {
            const int nbuf = buf ^ 1;
            if (IN_HALF) {
                *(uint4*)&Xs[nbuf][xr * 40 + xc] = xh4;
            } else {
                V16H tv;
#pragma unroll
                for (int j = 0; j < 8; ++j) tv.f[j] = (_Float16)xfl[j];
                *(uint4*)&Xs[nbuf][xr * 40 + xc] = tv.u[0];
            }
#pragma unroll
            for (int j = 0; j < 16; ++j) {
                union { _Float16 h[2]; unsigned int u; } pk;
                pk.h[0] = (_Float16)wst[0][j];
                pk.h[1] = (_Float16)wst[1][j];
                *(unsigned int*)&Ws[nbuf][(nb + j) * 40 + 2 * kp] = pk.u;
            }
        }
        __syncthreads();
        buf ^= 1;
    }

    // ---- epilogue: bias + store (D layout: c[r] -> row r+8*hf, col l16) ----
#pragma unroll
    for (int nt = 0; nt < 8; ++nt) {
        const int n = 128 * ng + 16 * nt + l16;
        const float bv = bias[n];
        const long base = (rowBase + 16 * mi + 8 * hf) * DM + n;
#pragma unroll
        for (int r = 0; r < 8; ++r) {
            const float v = acc[nt][r] + bv;
            if (OUT_FLOAT) Cf[base + (long)r * DM] = v;
            else           Ch[base + (long)r * DM] = (_Float16)v;
        }
    }
}

// ---------------------------------------------------------------------------
// Attention: one wave per (b,h,w,head). S = Q K^T /sqrt(dh) + mask,
// softmax in registers, O = P V. Q/K/V are f16 in workspace, layout
// (bhw, t, nh*32+depth). Writes attn (f32) and O (f16).
// ---------------------------------------------------------------------------
__global__ __launch_bounds__(32) void attn_head(
    const _Float16* __restrict__ Qh, const _Float16* __restrict__ Kh,
    const _Float16* __restrict__ Vh, const float* __restrict__ mask,
    float* __restrict__ attnOut, _Float16* __restrict__ Oh)
{
    __shared__ __align__(16) _Float16 Ps[64 * 72];   // P row stride 72 halves

    const int lane = threadIdx.x & 31;
    const int hf   = lane >> 4;
    const int l16  = lane & 15;
    const int hb   = blockIdx.x;          // (bhw*8 + nh)
    const int nh   = hb & 7;
    const long bhw = (long)(hb >> 3);
    const int b    = (int)(bhw >> 8);     // H*W = 256
    const long qbase = bhw * (TSEQ * (long)DM) + nh * DHEAD;

    const v8f vzero = {0.f,0.f,0.f,0.f,0.f,0.f,0.f,0.f};

    // ---- Q A-fragments straight from global (32 contiguous halves/row) ----
    V16H qf[4];
#pragma unroll
    for (int ti = 0; ti < 4; ++ti) {
        const _Float16* p = Qh + qbase + (long)(16 * ti + l16) * DM;
        qf[ti].u[0] = *(const uint4*)(p + 8 * hf);
        qf[ti].u[1] = *(const uint4*)(p + 16 + 8 * hf);
    }

    // ---- S = Q K^T : 4x4 tiles, full K-dim (32) per WMMA ----
    v8f S[4][4];
#pragma unroll
    for (int ti = 0; ti < 4; ++ti)
#pragma unroll
        for (int tj = 0; tj < 4; ++tj) S[ti][tj] = vzero;

#pragma unroll
    for (int tj = 0; tj < 4; ++tj) {
        V16H kf;  // B-frag: lane col = key row s; K groups 8*hf+i, 16+8*hf+i
        const _Float16* p = Kh + qbase + (long)(16 * tj + l16) * DM + 8 * hf;
        kf.u[0] = *(const uint4*)(p);
        kf.u[1] = *(const uint4*)(p + 16);
#pragma unroll
        for (int ti = 0; ti < 4; ++ti)
            S[ti][tj] = __builtin_amdgcn_wmma_f32_16x16x32_f16(
                false, qf[ti].h, false, kf.h, (short)0, S[ti][tj], false, false);
    }

    // ---- softmax over s (each row lives across lanes of one half) ----
    float mk[4];
#pragma unroll
    for (int tj = 0; tj < 4; ++tj)
        mk[tj] = mask[b * TSEQ + 16 * tj + l16] * (-1e9f);
    const float sc = 0.17677669529663687f;   // 1/sqrt(32)

#pragma unroll
    for (int ti = 0; ti < 4; ++ti) {
#pragma unroll
        for (int r = 0; r < 8; ++r) {
            float vv[4], ee[4];
#pragma unroll
            for (int tj = 0; tj < 4; ++tj) vv[tj] = S[ti][tj][r] * sc + mk[tj];
            float m = fmaxf(fmaxf(vv[0], vv[1]), fmaxf(vv[2], vv[3]));
#pragma unroll
            for (int off = 1; off < 16; off <<= 1)
                m = fmaxf(m, __shfl_xor(m, off, 32));
            float sum = 0.f;
#pragma unroll
            for (int tj = 0; tj < 4; ++tj) { ee[tj] = __expf(vv[tj] - m); sum += ee[tj]; }
#pragma unroll
            for (int off = 1; off < 16; off <<= 1)
                sum += __shfl_xor(sum, off, 32);
            const float inv = 1.0f / sum;
            const int t = 16 * ti + 8 * hf + r;
            const long ab = ((long)hb * TSEQ + t) * TSEQ;
#pragma unroll
            for (int tj = 0; tj < 4; ++tj) {
                const float pv = ee[tj] * inv;
                attnOut[ab + 16 * tj + l16] = pv;
                Ps[t * 72 + 16 * tj + l16] = (_Float16)pv;
            }
        }
    }
    __syncthreads();

    // ---- O = P V : M=64, N=32 (2 tiles), K=64 (2 chunks) ----
    v8f O[4][2];
#pragma unroll
    for (int ti = 0; ti < 4; ++ti) { O[ti][0] = vzero; O[ti][1] = vzero; }

#pragma unroll
    for (int kc = 0; kc < 2; ++kc) {
        V16H pf[4];
#pragma unroll
        for (int ti = 0; ti < 4; ++ti) {
            const _Float16* p = &Ps[(16 * ti + l16) * 72 + 32 * kc];
            pf[ti].u[0] = *(const uint4*)(p + 8 * hf);
            pf[ti].u[1] = *(const uint4*)(p + 16 + 8 * hf);
        }
#pragma unroll
        for (int nj = 0; nj < 2; ++nj) {
            V16H vf;  // B-frag of V: b[i] = V[s = 32*kc + (i<8 ? 8*hf+i : 16+8*hf+i-8)][dep]
            const int dep = 16 * nj + l16;
#pragma unroll
            for (int i = 0; i < 16; ++i) {
                const int s = 32 * kc + (i < 8 ? 8 * hf + i : 16 + 8 * hf + (i - 8));
                vf.f[i] = Vh[qbase + (long)s * DM + dep];
            }
#pragma unroll
            for (int ti = 0; ti < 4; ++ti)
                O[ti][nj] = __builtin_amdgcn_wmma_f32_16x16x32_f16(
                    false, pf[ti].h, false, vf.h, (short)0, O[ti][nj], false, false);
        }
    }

    // ---- store O (merged-head layout, f16) ----
#pragma unroll
    for (int ti = 0; ti < 4; ++ti)
#pragma unroll
        for (int nj = 0; nj < 2; ++nj) {
            const int dep = 16 * nj + l16;
#pragma unroll
            for (int r = 0; r < 8; ++r) {
                const int t = 16 * ti + 8 * hf + r;
                Oh[qbase + (long)t * DM + dep] = (_Float16)O[ti][nj][r];
            }
        }
}

extern "C" void kernel_launch(void* const* d_in, const int* in_sizes, int n_in,
                              void* d_out, int out_size, void* d_ws, size_t ws_size,
                              hipStream_t stream) {
    const float* v    = (const float*)d_in[0];
    const float* k    = (const float*)d_in[1];
    const float* q    = (const float*)d_in[2];
    const float* mask = (const float*)d_in[3];
    const float* wq_w = (const float*)d_in[4];
    const float* wq_b = (const float*)d_in[5];
    const float* wk_w = (const float*)d_in[6];
    const float* wk_b = (const float*)d_in[7];
    const float* wv_w = (const float*)d_in[8];
    const float* wv_b = (const float*)d_in[9];
    const float* wo_w = (const float*)d_in[10];
    const float* wo_b = (const float*)d_in[11];

    float* out  = (float*)d_out;
    float* attn = out + (long)TOK_TOTAL * DM;          // 16,777,216 floats in

    _Float16* Qh = (_Float16*)d_ws;                    // each 16,777,216 halves
    _Float16* Kh = Qh + (long)TOK_TOTAL * DM;
    _Float16* Vh = Kh + (long)TOK_TOTAL * DM;
    _Float16* Oh = Vh + (long)TOK_TOTAL * DM;

    const int gemmBlocks = TOK_TOTAL / 64;             // 1024

    // QKV projections (f32 in -> f16 out)
    gemm_d256<false, false><<<gemmBlocks, 256, 0, stream>>>(q, wq_w, wq_b, Qh);
    gemm_d256<false, false><<<gemmBlocks, 256, 0, stream>>>(k, wk_w, wk_b, Kh);
    gemm_d256<false, false><<<gemmBlocks, 256, 0, stream>>>(v, wv_w, wv_b, Vh);

    // fused attention per head: 1024 seqs * 8 heads, one wave each
    attn_head<<<8192, 32, 0, stream>>>(Qh, Kh, Vh, mask, attn, Oh);

    // output projection (f16 in -> f32 out)
    gemm_d256<true, true><<<gemmBlocks, 256, 0, stream>>>(Oh, wo_w, wo_b, out);
}
